// NeuralDecisionTree_37615323578591
// MI455X (gfx1250) — matrix operand order repaired and checked by reading, not compiled, not run on hardware
//
#include <hip/hip_runtime.h>
#include <hip/hip_bf16.h>

// ---------------------------------------------------------------------------
// Neural decision tree: out = kron_leaf(x, cut_points) @ leaf_score
//   BATCH=2048, NUM_FEATS=8, NUM_CUTS=3 (-> 4 bins/feat), NUM_LEAF=4^8=65536,
//   NUM_CLASSES=10.
// Strategy: never materialize the (2048 x 65536) leaf matrix. leaf[b,l] is a
// product of 8 per-feature factors; build A fragments in registers from
// prefix/suffix products and contract with bf16 WMMA (v_wmma_f32_16x16x32_bf16).
// leaf_score is pre-converted once into WMMA B-fragment layout in d_ws (2 MB,
// L2-resident and reused by every block). Dual accumulators break the WMMA
// C->D dependency chain so VALU A-builds co-execute in the XDL shadow.
// ---------------------------------------------------------------------------

typedef __attribute__((ext_vector_type(16))) __bf16 v16bf;
typedef __attribute__((ext_vector_type(8)))  float  v8f;

#define NDT_BATCH   2048
#define NDT_NFEAT   8
#define NDT_NCLS    10
#define NDT_NCHUNK  2048   // 65536 / 32 K-chunks
#define NDT_KSPLIT  4
#define NDT_NTILE   128    // 2048 / 16 batch tiles

// ---------------------------------------------------------------------------
// Prep: leaf_score (65536 x 10, f32) -> packed bf16 B fragments in ws.
// B layout for V_WMMA_F32_16X16X32_BF16 (B is 32x16, 8 VGPRs):
//   lane L: column n = L&15 ; lanes 0-15 hold K=0..7,16..23, lanes 16-31 K=8..15,24..31
//   VGPR v holds packed pair (K = base+2v) in [15:0], (base+2v+1) in [31:16]
// Stored as wsB[c*256 + L*8 + v] so each lane loads 8 consecutive dwords.
// ---------------------------------------------------------------------------
__global__ __launch_bounds__(256)
void ndt_prep_b(const float* __restrict__ ls, unsigned int* __restrict__ wsB) {
  unsigned t    = blockIdx.x * 256u + threadIdx.x;   // 0 .. 524287
  unsigned c    = t >> 8;
  unsigned r    = t & 255u;
  unsigned L    = r >> 3;
  unsigned v    = r & 7u;
  unsigned n    = L & 15u;
  unsigned half = (L >> 4) & 1u;
  unsigned k    = c * 32u + (v < 4u ? 0u : 16u) + (v & 3u) * 2u + half * 8u;
  float f0 = (n < NDT_NCLS) ? ls[(size_t)k * NDT_NCLS + n]       : 0.0f;
  float f1 = (n < NDT_NCLS) ? ls[(size_t)(k + 1) * NDT_NCLS + n] : 0.0f;
  __bf16 b0 = (__bf16)f0;
  __bf16 b1 = (__bf16)f1;
  unsigned u0 = (unsigned)__builtin_bit_cast(unsigned short, b0);
  unsigned u1 = (unsigned)__builtin_bit_cast(unsigned short, b1);
  wsB[(size_t)c * 256u + L * 8u + v] = u0 | (u1 << 16);
}

// one K=32 chunk: build A fragment from prefix p and register suffix pairs,
// load pre-swizzled B, accumulate with bf16 WMMA into the given accumulator.
#define NDT_CHUNK(SVARR, ACC)                                                 \
  {                                                                           \
    float p = PAm[c >> 5] * PBm[(c >> 1) & 15u];                              \
    v16bf a;                                                                  \
    _Pragma("unroll")                                                         \
    for (int v = 0; v < 8; ++v) {                                             \
      a[2 * v]     = (__bf16)(p * SVARR[v].x);                                \
      a[2 * v + 1] = (__bf16)(p * SVARR[v].y);                                \
    }                                                                         \
    const uint4* bp = (const uint4*)(wsB + (size_t)c * 256u + lane * 8u);     \
    union BU { uint4 u[2]; v16bf bv; } bu;                                    \
    bu.u[0] = bp[0];                                                          \
    bu.u[1] = bp[1];                                                          \
    ACC = __builtin_amdgcn_wmma_f32_16x16x32_bf16(false, a, false, bu.bv,     \
                                                  (short)0, ACC, false,       \
                                                  false);                     \
  }

// ---------------------------------------------------------------------------
// Main: block = (batch tile of 16 rows) x (K split of 512 chunks), 8 waves.
// ---------------------------------------------------------------------------
__global__ __launch_bounds__(256)
void ndt_main(const float* __restrict__ x, const float* __restrict__ cp,
              const unsigned int* __restrict__ wsB,
              float* __restrict__ wsPart) {
  __shared__ __align__(16) float bias_lds[32];      // [feat][bin]
  __shared__ __align__(16) float h_lds[16 * 8 * 4]; // [m][feat][bin]
  __shared__ __align__(16) float S_lds[16 * 64];    // suffix: feats 5,6,7
  __shared__ __align__(16) float PA_lds[16 * 64];   // prefix hi: feats 0,1,2
  __shared__ __align__(16) float PB_lds[16 * 16];   // prefix lo: feats 3,4
  __shared__ __align__(16) float part_lds[8][256];  // per-wave partial C

  const unsigned tid  = threadIdx.x;
  const unsigned tile = blockIdx.x / NDT_KSPLIT;
  const unsigned ks   = blockIdx.x % NDT_KSPLIT;

  // --- biases: b[f][k] = -(sum of first k sorted cuts) -----------------------
  if (tid < 32) {
    unsigned f = tid >> 2, kk = tid & 3u;
    float c0 = cp[f * 3 + 0], c1 = cp[f * 3 + 1], c2 = cp[f * 3 + 2];
    float lo01 = fminf(c0, c1), hi01 = fmaxf(c0, c1);
    float s0 = fminf(lo01, c2);
    float s2 = fmaxf(hi01, c2);
    float s1 = (c0 + c1 + c2) - s0 - s2;
    float b = 0.0f;
    if (kk == 1) b = -s0;
    else if (kk == 2) b = -(s0 + s1);
    else if (kk == 3) b = -(s0 + s1 + s2);
    bias_lds[f * 4 + kk] = b;
  }
  __syncthreads();

  // --- h[m][f][d] = x * (d+1) + bias ----------------------------------------
  if (tid < 128) {
    unsigned m = tid >> 3, f = tid & 7u;
    float xv = x[((size_t)tile * 16 + m) * NDT_NFEAT + f];
#pragma unroll
    for (int d = 0; d < 4; ++d)
      h_lds[(m * 8 + f) * 4 + d] = xv * (float)(d + 1) + bias_lds[f * 4 + d];
  }
  __syncthreads();

  // --- Kronecker factors: S (feats 5..7), PA (feats 0..2), PB (feats 3..4) --
#pragma unroll
  for (int e = 0; e < 4; ++e) {
    unsigned idx = tid * 4 + e;           // 0 .. 1023
    unsigned m = idx >> 6, q = idx & 63u;
    unsigned da = q >> 4, db = (q >> 2) & 3u, dc = q & 3u;
    const float* hm = &h_lds[m * 32];
    S_lds[idx]  = hm[5 * 4 + da] * hm[6 * 4 + db] * hm[7 * 4 + dc];
    PA_lds[idx] = hm[0 * 4 + da] * hm[1 * 4 + db] * hm[2 * 4 + dc];
  }
  {
    unsigned m = tid >> 4, d3 = (tid >> 2) & 3u, d4 = tid & 3u;
    const float* hm = &h_lds[m * 32];
    PB_lds[tid] = hm[3 * 4 + d3] * hm[4 * 4 + d4];
  }
  __syncthreads();

  // --- hoist this lane's suffix values into registers ------------------------
  // A layout (16x32 bf16): lane L -> row m = L&15; VGPR v holds packed K pair
  //   K = (v<4 ? 0 : 16) + (v&3)*2 + (L>=16 ? 8 : 0)
  const unsigned lane = tid & 31u;
  const unsigned wv   = tid >> 5;      // wave 0..7
  const unsigned m    = lane & 15u;    // batch row of this lane's A elements
  const unsigned half = (lane >> 4) & 1u;

  float2 sv0[8], sv1[8];
#pragma unroll
  for (int v = 0; v < 8; ++v) {
    unsigned k0 = (v < 4 ? 0u : 16u) + (unsigned)(v & 3) * 2u + half * 8u;
    sv0[v] = *(const float2*)&S_lds[m * 64 + k0];        // chunk parity 0
    sv1[v] = *(const float2*)&S_lds[m * 64 + 32 + k0];   // chunk parity 1
  }

  const float* PAm = &PA_lds[m * 64];
  const float* PBm = &PB_lds[m * 16];

  // Independent accumulators per parity: breaks the WMMA C->D serial chain so
  // the next chunk's A-build VALU ops co-execute in the XDL shadow (no v_nops).
  v8f acc0 = {};
  v8f acc1 = {};
  const unsigned span = NDT_NCHUNK / NDT_KSPLIT / 8;     // 64 chunks per wave
  const unsigned cbeg = ks * (NDT_NCHUNK / NDT_KSPLIT) + wv * span;
  const unsigned cend = cbeg + span;

#pragma unroll 2
  for (unsigned c = cbeg; c < cend;) {
    NDT_CHUNK(sv0, acc0); ++c;   // even chunk: low half of S block
    NDT_CHUNK(sv1, acc1); ++c;   // odd chunk: high half
  }

  v8f acc;
#pragma unroll
  for (int r = 0; r < 8; ++r) acc[r] = acc0[r] + acc1[r];

  // --- deterministic cross-wave reduction via LDS ---------------------------
  // C layout: lane L -> col n = L&15 ; VGPR r -> row r + (L>=16 ? 8 : 0)
#pragma unroll
  for (int r = 0; r < 8; ++r)
    part_lds[wv][(r + half * 8) * 16 + m] = acc[r];
  __syncthreads();

  float s = 0.0f;
#pragma unroll
  for (int w = 0; w < 8; ++w) s += part_lds[w][tid];
  wsPart[(size_t)blockIdx.x * 256 + tid] = s;
}

// ---------------------------------------------------------------------------
// Final reduce over the K split, write 2048 x 10 f32.
// ---------------------------------------------------------------------------
__global__ __launch_bounds__(256)
void ndt_reduce(const float* __restrict__ wsPart, float* __restrict__ out) {
  unsigned t = blockIdx.x * 256u + threadIdx.x;
  if (t >= NDT_BATCH * NDT_NCLS) return;
  unsigned brow = t / NDT_NCLS, ncls = t % NDT_NCLS;
  unsigned tile = brow >> 4, mm = brow & 15u;
  float s = 0.0f;
#pragma unroll
  for (int ksi = 0; ksi < NDT_KSPLIT; ++ksi)
    s += wsPart[(size_t)(tile * NDT_KSPLIT + ksi) * 256 + mm * 16 + ncls];
  out[t] = s;
}

extern "C" void kernel_launch(void* const* d_in, const int* in_sizes, int n_in,
                              void* d_out, int out_size, void* d_ws,
                              size_t ws_size, hipStream_t stream) {
  (void)in_sizes; (void)n_in; (void)out_size; (void)ws_size;
  const float* x  = (const float*)d_in[0];   // 2048 x 8
  const float* cp = (const float*)d_in[1];   // 8 x 3
  const float* ls = (const float*)d_in[2];   // 65536 x 10
  float* out = (float*)d_out;                // 2048 x 10

  // ws: [0, 2MB) packed bf16 B fragments; [2MB, 2.5MB) K-split partials
  unsigned int* wsB = (unsigned int*)d_ws;
  float* wsPart = (float*)((char*)d_ws + (size_t)NDT_NCHUNK * 256 * 4);

  ndt_prep_b<<<NDT_NCHUNK, 256, 0, stream>>>(ls, wsB);
  ndt_main<<<NDT_NTILE * NDT_KSPLIT, 256, 0, stream>>>(x, cp, wsB, wsPart);
  ndt_reduce<<<(NDT_BATCH * NDT_NCLS + 255) / 256, 256, 0, stream>>>(wsPart, out);
}
